// BERT_GAT_39951785787636
// MI455X (gfx1250) — matrix operand order repaired and checked by reading, not compiled, not run on hardware
//
#include <hip/hip_runtime.h>

// ---------------------------------------------------------------------------
// GAT 5-layer pipeline for gfx1250 (MI455X).
// GEMM: v_wmma_f32_16x16x32_bf16, fp32 accumulate, 128x128x32 block tile,
//       8 WMMAs / wave / K-step, register-staged software pipeline,
//       fully padded (no bounds checks anywhere in the hot path).
// Edge path: atomics-based segment softmax + scatter (bandwidth-trivial).
// ---------------------------------------------------------------------------

typedef __attribute__((ext_vector_type(16))) __bf16 v16bf;
typedef __attribute__((ext_vector_type(8)))  __bf16 v8bf;
typedef __attribute__((ext_vector_type(8)))  float  v8f;

#define GAT_N  50000
#define GAT_E  250000
#define GAT_E2 (GAT_E + GAT_N)   // with self loops
#define NEG_SLOPE 0.2f

#define BM 128
#define BN 128
#define BK 32
#define MPAD  50048              // 391 * 128
#define K0PAD 224                // layer-0 K (200) padded to BK multiple
#define LDSP  (BK + 8)           // LDS row stride (elems): 80B, 16B-aligned

// ------------------- GEMM: C = A x B (bf16 WMMA, all dims padded) ----------
// lda, ldb, ldc in elements; K % 32 == 0; grid covers padded M, N exactly.
// 8 waves: 4 along M x 2 along N; each wave computes a 32x64 tile (2x4 WMMA).
__global__ __launch_bounds__(256)
void gemm_bf16_wmma(const float* __restrict__ A, int lda,
                    const float* __restrict__ B, int ldb,
                    float* __restrict__ C, int ldc, int K)
{
    __shared__ __align__(16) __bf16 As[BM][LDSP];   // 10 KB
    __shared__ __align__(16) __bf16 Bt[BN][LDSP];   // 10 KB, transposed [n][k]

    const int tid  = threadIdx.x;
    const int wave = tid >> 5;
    const int lane = tid & 31;
    const int wm   = wave >> 1;          // 0..3  -> 32 rows
    const int wn   = wave & 1;           // 0..1  -> 64 cols
    const long m0  = (long)blockIdx.y * BM;
    const long n0  = (long)blockIdx.x * BN;

    const int l15   = lane & 15;
    const int hi16  = lane >> 4;
    const int kselA = hi16 * 8;          // A frag K base: 0 or 8
    const int kselB = hi16 * 8;          // B frag K base: 0 or 8

    // staging assignments (vectorized, unconditional)
    const int ar = tid >> 2;             // 0..63 (and +64)
    const int ak = (tid & 3) * 8;        // 0,8,16,24
    const int bk = tid >> 4;             // 0..15 (and +16)
    const int bn = (tid & 15) * 8;       // 0..120

    const float* gA0 = A + (m0 + ar) * lda + ak;
    const float* gA1 = A + (m0 + ar + 64) * lda + ak;
    const float* gB0 = B + (long)bk * ldb + (n0 + bn);
    const float* gB1 = B + (long)(bk + 16) * ldb + (n0 + bn);

    float ra0[8], ra1[8], rb0[8], rb1[8];
    auto preload = [&](int kc) {
        const float4* p = (const float4*)(gA0 + kc);
        float4 t0 = p[0], t1 = p[1];
        ra0[0]=t0.x; ra0[1]=t0.y; ra0[2]=t0.z; ra0[3]=t0.w;
        ra0[4]=t1.x; ra0[5]=t1.y; ra0[6]=t1.z; ra0[7]=t1.w;
        p = (const float4*)(gA1 + kc);
        t0 = p[0]; t1 = p[1];
        ra1[0]=t0.x; ra1[1]=t0.y; ra1[2]=t0.z; ra1[3]=t0.w;
        ra1[4]=t1.x; ra1[5]=t1.y; ra1[6]=t1.z; ra1[7]=t1.w;
        p = (const float4*)(gB0 + (long)kc * ldb);
        t0 = p[0]; t1 = p[1];
        rb0[0]=t0.x; rb0[1]=t0.y; rb0[2]=t0.z; rb0[3]=t0.w;
        rb0[4]=t1.x; rb0[5]=t1.y; rb0[6]=t1.z; rb0[7]=t1.w;
        p = (const float4*)(gB1 + (long)kc * ldb);
        t0 = p[0]; t1 = p[1];
        rb1[0]=t0.x; rb1[1]=t0.y; rb1[2]=t0.z; rb1[3]=t0.w;
        rb1[4]=t1.x; rb1[5]=t1.y; rb1[6]=t1.z; rb1[7]=t1.w;
    };

    v8f acc[2][4] = {};
    preload(0);

    for (int k0 = 0; k0 < K; k0 += BK) {
        // ---- stage registers -> LDS (convert to bf16) ----
        {
            v8bf t;
            #pragma unroll
            for (int i = 0; i < 8; ++i) t[i] = (__bf16)ra0[i];
            *(v8bf*)&As[ar][ak] = t;
            #pragma unroll
            for (int i = 0; i < 8; ++i) t[i] = (__bf16)ra1[i];
            *(v8bf*)&As[ar + 64][ak] = t;
            #pragma unroll
            for (int i = 0; i < 8; ++i) Bt[bn + i][bk] = (__bf16)rb0[i];
            #pragma unroll
            for (int i = 0; i < 8; ++i) Bt[bn + i][bk + 16] = (__bf16)rb1[i];
        }
        __syncthreads();

        // ---- preload next K-tile (overlaps with ds_load + wmma below) ----
        int kn = k0 + BK;
        if (kn >= K) kn = 0;
        preload(kn);

        // ---- fragments (CDNA5 wave32 WMMA register layouts) ----
        v16bf afrag[2], bfrag[4];
        #pragma unroll
        for (int mi = 0; mi < 2; ++mi) {
            const __bf16* ap = &As[wm * 32 + mi * 16 + l15][0];
            v8bf lo = *(const v8bf*)(ap + kselA);        // K = ksel..+7
            v8bf hh = *(const v8bf*)(ap + kselA + 16);   // K = ksel+16..+23
            v16bf a;
            #pragma unroll
            for (int i = 0; i < 8; ++i) { a[i] = lo[i]; a[i + 8] = hh[i]; }
            afrag[mi] = a;
        }
        #pragma unroll
        for (int ni = 0; ni < 4; ++ni) {
            const __bf16* bp = &Bt[wn * 64 + ni * 16 + l15][0];
            v8bf lo = *(const v8bf*)(bp + kselB);        // K = ksel..+7
            v8bf hh = *(const v8bf*)(bp + kselB + 16);   // K = ksel+16..+23
            v16bf b;
            #pragma unroll
            for (int i = 0; i < 8; ++i) { b[i] = lo[i]; b[i + 8] = hh[i]; }
            bfrag[ni] = b;
        }

        #pragma unroll
        for (int mi = 0; mi < 2; ++mi)
            #pragma unroll
            for (int ni = 0; ni < 4; ++ni)
                acc[mi][ni] = __builtin_amdgcn_wmma_f32_16x16x32_bf16(
                    false, afrag[mi], false, bfrag[ni],
                    (short)0, acc[mi][ni], false, false);

        __syncthreads();
    }

    // ---- store (no bounds checks: C is padded) ----
    #pragma unroll
    for (int mi = 0; mi < 2; ++mi) {
        #pragma unroll
        for (int ni = 0; ni < 4; ++ni) {
            long col = n0 + wn * 64 + ni * 16 + l15;
            long row = m0 + wm * 32 + mi * 16 + hi16 * 8;
            float* cp = C + row * ldc + col;
            #pragma unroll
            for (int v = 0; v < 8; ++v) cp[(long)v * ldc] = acc[mi][ni][v];
        }
    }
}

// ------------------------- helpers -----------------------------------------
__device__ __forceinline__ unsigned f2ord(float f) {
    unsigned u = __float_as_uint(f);
    return (u & 0x80000000u) ? ~u : (u | 0x80000000u);
}
__device__ __forceinline__ float ord2f(unsigned o) {
    unsigned u = (o & 0x80000000u) ? (o & 0x7fffffffu) : ~o;
    return __uint_as_float(u);
}
__device__ __forceinline__ float gelu_tanh(float x) {
    float x3 = x * x * x;
    return 0.5f * x * (1.0f + tanhf(0.7978845608028654f * (x + 0.044715f * x3)));
}

// ------------------------- padded copies -----------------------------------
__global__ void pad_copy_x0(const float* __restrict__ X, float* __restrict__ x0)
{
    long i = (long)blockIdx.x * blockDim.x + threadIdx.x;
    if (i >= (long)MPAD * K0PAD) return;
    int r = (int)(i / K0PAD), k = (int)(i - (long)r * K0PAD);
    x0[i] = (r < GAT_N && k < 200) ? X[(long)r * 200 + k] : 0.0f;
}

__global__ void pad_copy_w(const float* __restrict__ W, float* __restrict__ wp,
                           int K, int hC, int Kpad, int hCpad)
{
    long i = (long)blockIdx.x * blockDim.x + threadIdx.x;
    if (i >= (long)Kpad * hCpad) return;
    int k = (int)(i / hCpad), n = (int)(i - (long)k * hCpad);
    wp[i] = (k < K && n < hC) ? W[(long)k * hC + n] : 0.0f;
}

// ------------------------- edge prep ---------------------------------------
__global__ void prep_edges(const long long* __restrict__ ei,
                           int* __restrict__ es, int* __restrict__ ed)
{
    int i = blockIdx.x * blockDim.x + threadIdx.x;
    if (i >= GAT_E2) return;
    if (i < GAT_E) { es[i] = (int)ei[i]; ed[i] = (int)ei[GAT_E + i]; }
    else           { es[i] = i - GAT_E;  ed[i] = i - GAT_E; }
}

// ------------------------- per-node attention coefficients -----------------
__global__ __launch_bounds__(128)
void attn_coef(const float* __restrict__ feat, int ldF,
               const float* __restrict__ a_src, const float* __restrict__ a_dst,
               float* __restrict__ als, float* __restrict__ ald, int h, int C)
{
    int n = blockIdx.x;
    int w = threadIdx.x >> 5;
    int lane = threadIdx.x & 31;
    if (w >= h) return;
    const float* f  = feat + (long)n * ldF + (long)w * C;
    const float* pa = a_src + w * C;
    const float* pd = a_dst + w * C;
    float ss = 0.f, sd = 0.f;
    for (int c = lane; c < C; c += 32) {
        float fv = f[c];
        ss += fv * pa[c];
        sd += fv * pd[c];
    }
    for (int off = 16; off; off >>= 1) {
        ss += __shfl_down(ss, off, 32);
        sd += __shfl_down(sd, off, 32);
    }
    if (lane == 0) { als[n * h + w] = ss; ald[n * h + w] = sd; }
}

// ------------------------- init / zero -------------------------------------
__global__ void init_ms(unsigned* __restrict__ mu, float* __restrict__ ssum, int n)
{
    int i = blockIdx.x * blockDim.x + threadIdx.x;
    if (i >= n) return;
    mu[i] = f2ord(-__builtin_inff());
    ssum[i] = 0.0f;
}

__global__ void zero_f32(float* __restrict__ p, long n)
{
    long i = (long)blockIdx.x * blockDim.x + threadIdx.x;
    if (i < n) p[i] = 0.0f;
}

// ------------------------- pass 1: logits + segment max --------------------
__global__ void edge_logit_max(const int* __restrict__ es, const int* __restrict__ ed,
                               const float* __restrict__ als, const float* __restrict__ ald,
                               float* __restrict__ ebuf, unsigned* __restrict__ mu, int h)
{
    int i = blockIdx.x * blockDim.x + threadIdx.x;
    if (i >= GAT_E2 * h) return;
    int e = i / h, hh = i - e * h;
    float v = als[es[e] * h + hh] + ald[ed[e] * h + hh];
    v = (v > 0.f) ? v : v * NEG_SLOPE;
    ebuf[i] = v;
    atomicMax(&mu[ed[e] * h + hh], f2ord(v));
}

// ------------------------- pass 2: exp + segment sum -----------------------
__global__ void edge_expsum(const int* __restrict__ ed,
                            const float* __restrict__ ebuf, const unsigned* __restrict__ mu,
                            float* __restrict__ pbuf, float* __restrict__ ssum, int h)
{
    int i = blockIdx.x * blockDim.x + threadIdx.x;
    if (i >= GAT_E2 * h) return;
    int e = i / h, hh = i - e * h;
    float m = ord2f(mu[ed[e] * h + hh]);
    if (!isfinite(m)) m = 0.0f;
    float p = __expf(ebuf[i] - m);
    pbuf[i] = p;
    atomicAdd(&ssum[ed[e] * h + hh], p);
}

// ------------------------- pass 3: weighted scatter-aggregate --------------
__global__ __launch_bounds__(256)
void edge_aggregate(const int* __restrict__ es, const int* __restrict__ ed,
                    const float* __restrict__ feat, int ldF,
                    const float* __restrict__ pbuf, const float* __restrict__ ssum,
                    float* __restrict__ out, int h, int C)
{
    int e = blockIdx.x;
    int s = es[e], d = ed[e];
    const float* fs = feat + (long)s * ldF;
    float* o = out + (long)d * h * C;
    for (int hh = 0; hh < h; ++hh) {
        float a = pbuf[e * h + hh] / (ssum[d * h + hh] + 1e-16f);
        for (int j = threadIdx.x; j < C; j += blockDim.x)
            atomicAdd(&o[hh * C + j], fs[hh * C + j] * a);
    }
}

// ------------------------- bias + gelu -------------------------------------
__global__ void bias_gelu(const float* __restrict__ x, const float* __restrict__ bias,
                          float* __restrict__ dst, long n, int hC)
{
    long i = (long)blockIdx.x * blockDim.x + threadIdx.x;
    if (i >= n) return;
    dst[i] = gelu_tanh(x[i] + bias[i % hC]);
}

// ------------------------- final-layer alpha (h==1) ------------------------
__global__ void alpha_out(const int* __restrict__ ed, const float* __restrict__ pbuf,
                          const float* __restrict__ ssum, float* __restrict__ dst)
{
    int i = blockIdx.x * blockDim.x + threadIdx.x;
    if (i >= GAT_E2) return;
    dst[i] = pbuf[i] / (ssum[ed[i]] + 1e-16f);
}

// ===========================================================================
extern "C" void kernel_launch(void* const* d_in, const int* in_sizes, int n_in,
                              void* d_out, int out_size, void* d_ws, size_t ws_size,
                              hipStream_t stream)
{
    const float*     X  = (const float*)d_in[0];
    const long long* EI = (const long long*)d_in[1];

    const float* Wl[5], *ASl[5], *ADl[5], *Bl[5];
    for (int l = 0; l < 5; ++l) {
        Wl[l]  = (const float*)d_in[2 + 4 * l];
        ASl[l] = (const float*)d_in[3 + 4 * l];
        ADl[l] = (const float*)d_in[4 + 4 * l];
        Bl[l]  = (const float*)d_in[5 + 4 * l];
    }

    auto align_up = [](size_t v) { return (v + 255) & ~(size_t)255; };
    char* ws = (char*)d_ws;
    size_t off = 0;
    int*      es   = (int*)(ws + off);      off += align_up((size_t)GAT_E2 * 4);
    int*      ed   = (int*)(ws + off);      off += align_up((size_t)GAT_E2 * 4);
    float*    x0   = (float*)(ws + off);    off += align_up((size_t)MPAD * K0PAD * 4);
    float*    wpad = (float*)(ws + off);    off += align_up((size_t)800 * 896 * 4);
    float*    feat = (float*)(ws + off);    off += align_up((size_t)MPAD * 896 * 4);
    float*    xbuf = (float*)(ws + off);    off += align_up((size_t)MPAD * 800 * 4);
    float*    als  = (float*)(ws + off);    off += align_up((size_t)GAT_N * 4 * 4);
    float*    ald  = (float*)(ws + off);    off += align_up((size_t)GAT_N * 4 * 4);
    unsigned* mu   = (unsigned*)(ws + off); off += align_up((size_t)GAT_N * 4 * 4);
    float*    ssum = (float*)(ws + off);    off += align_up((size_t)GAT_N * 4 * 4);
    float*    ebuf = (float*)(ws + off);    off += align_up((size_t)GAT_E2 * 4 * 4);
    float*    pbuf = (float*)(ws + off);    off += align_up((size_t)GAT_E2 * 4 * 4);
    (void)ws_size; (void)in_sizes; (void)n_in; (void)out_size;

    float* out_x     = (float*)d_out;                        // [N, 200]
    float* out_alpha = (float*)d_out + (size_t)GAT_N * 200;  // [E2]

    prep_edges<<<(GAT_E2 + 255) / 256, 256, 0, stream>>>(EI, es, ed);

    // layer-0 activation, zero-padded to [MPAD x K0PAD]
    {
        long n = (long)MPAD * K0PAD;
        pad_copy_x0<<<(n + 255) / 256, 256, 0, stream>>>(X, x0);
    }
    // zero pad-rows of xbuf once (rows >= N are never rewritten)
    {
        long n = (long)(MPAD - GAT_N) * 800;
        zero_f32<<<(n + 255) / 256, 256, 0, stream>>>(xbuf + (long)GAT_N * 800, n);
    }

    const int Fin[5]  = {K0PAD, 800, 800, 800, 800};   // padded K
    const int Hh[5]   = {4, 4, 4, 4, 1};
    const int HCp[5]  = {896, 896, 896, 896, 256};     // padded output cols (BN mult)
    const int FinR[5] = {200, 800, 800, 800, 800};     // real K
    const int Cc      = 200;

    for (int l = 0; l < 5; ++l) {
        const int Kp   = Fin[l];
        const int h    = Hh[l];
        const int hC   = h * Cc;
        const int hCp  = HCp[l];
        const float* xin = (l == 0) ? x0 : xbuf;
        const int lda    = (l == 0) ? K0PAD : 800;

        // weights, zero-padded to [Kp x hCp]
        {
            long n = (long)Kp * hCp;
            pad_copy_w<<<(n + 255) / 256, 256, 0, stream>>>(Wl[l], wpad,
                                                            FinR[l], hC, Kp, hCp);
        }

        // feat = xin @ W  (bf16 WMMA, fp32 accumulate; fully padded)
        dim3 ggrid(hCp / BN, MPAD / BM);
        gemm_bf16_wmma<<<ggrid, 256, 0, stream>>>(xin, lda, wpad, hCp,
                                                  feat, hCp, Kp);

        // per-node attention coefficients
        attn_coef<<<GAT_N, 128, 0, stream>>>(feat, hCp, ASl[l], ADl[l],
                                             als, ald, h, Cc);

        // segment softmax over incoming edges
        init_ms<<<(GAT_N * h + 255) / 256, 256, 0, stream>>>(mu, ssum, GAT_N * h);
        edge_logit_max<<<((long)GAT_E2 * h + 255) / 256, 256, 0, stream>>>(
            es, ed, als, ald, ebuf, mu, h);
        edge_expsum<<<((long)GAT_E2 * h + 255) / 256, 256, 0, stream>>>(
            ed, ebuf, mu, pbuf, ssum, h);

        // weighted scatter-aggregate into xbuf (packed [N, hC])
        long nxt = (long)GAT_N * hC;
        zero_f32<<<(nxt + 255) / 256, 256, 0, stream>>>(xbuf, nxt);
        edge_aggregate<<<GAT_E2, 256, 0, stream>>>(es, ed, feat, hCp,
                                                   pbuf, ssum, xbuf, h, Cc);

        // bias + gelu (last layer writes straight into d_out)
        float* dst = (l == 4) ? out_x : xbuf;
        bias_gelu<<<(nxt + 255) / 256, 256, 0, stream>>>(xbuf, Bl[l], dst, nxt, hC);
    }

    // final-layer alpha (h == 1)
    alpha_out<<<(GAT_E2 + 255) / 256, 256, 0, stream>>>(ed, pbuf, ssum, out_alpha);
}
